// DGL_GCN_30949534335205
// MI455X (gfx1250) — matrix operand order repaired and checked by reading, not compiled, not run on hardware
//
#include <hip/hip_runtime.h>

// ---------------------------------------------------------------------------
// Types for gfx1250 WMMA (wave32): D(16x16 f32) = A(16x32 bf16) * B(32x16 bf16) + C
// ---------------------------------------------------------------------------
typedef __attribute__((ext_vector_type(16))) __bf16 v16bf;
typedef __attribute__((ext_vector_type(8)))  float  v8f;

union BfFrag { unsigned u[8]; v16bf v; };

__device__ __forceinline__ unsigned short f2bf(float f) {
    unsigned u = __float_as_uint(f);
    u += 0x7fffu + ((u >> 16) & 1u);          // round-to-nearest-even
    return (unsigned short)(u >> 16);
}
__device__ __forceinline__ float bf2f(unsigned short h) {
    return __uint_as_float(((unsigned)h) << 16);
}
__device__ __forceinline__ unsigned pk2(float lo, float hi) {
    return (unsigned)f2bf(lo) | ((unsigned)f2bf(hi) << 16);
}

// ---------------------------------------------------------------------------
// Degree accumulation + symmetric 'both' norm
// ---------------------------------------------------------------------------
__global__ void deg_kernel(const int* __restrict__ src, const int* __restrict__ dst,
                           float* __restrict__ degS, float* __restrict__ degD, int E) {
    int i = blockIdx.x * blockDim.x + threadIdx.x;
    if (i < E) {
        atomicAdd(&degS[src[i]], 1.0f);
        atomicAdd(&degD[dst[i]], 1.0f);
    }
}

__global__ void norm_kernel(float* __restrict__ nS, float* __restrict__ nD, int N) {
    int i = blockIdx.x * blockDim.x + threadIdx.x;
    if (i < N) {
        float a = nS[i], b = nD[i];
        nS[i] = (a > 0.0f) ? rsqrtf(a) : 0.0f;
        nD[i] = (b > 0.0f) ? rsqrtf(b) : 0.0f;
    }
}

// ---------------------------------------------------------------------------
// out[j] = dot(v, W[j, :]) + bias[j]   (one block per output row, LDS reduce)
// Used for ce = class_embed @ W_img.T and final o @ W_fin.T
// ---------------------------------------------------------------------------
__global__ void __launch_bounds__(256)
vecmat_kernel(const float* __restrict__ v, const float* __restrict__ W,
              const float* __restrict__ bias, float* __restrict__ out, int N) {
    __shared__ float red[256];
    const int j = blockIdx.x;
    const float* row = W + (size_t)j * N;
    float s = 0.0f;
    for (int i = threadIdx.x; i < N; i += blockDim.x) s += v[i] * row[i];
    red[threadIdx.x] = s;
    __syncthreads();
    for (int st = 128; st > 0; st >>= 1) {
        if ((int)threadIdx.x < st) red[threadIdx.x] += red[threadIdx.x + st];
        __syncthreads();
    }
    if (threadIdx.x == 0) out[j] = red[0] + bias[j];
}

// ---------------------------------------------------------------------------
// f32 [rows, cols] -> bf16 [rows, ldout] (pad cols assumed pre-zeroed)
// ---------------------------------------------------------------------------
__global__ void cvt_pad_kernel(const float* __restrict__ in, unsigned short* __restrict__ out,
                               int cols, int ldout) {
    const int r = blockIdx.x;
    for (int c = threadIdx.x; c < cols; c += blockDim.x)
        out[(size_t)r * ldout + c] = f2bf(in[(size_t)r * cols + c]);
}

// Transpose-convert: Wt[j, k] = bf16(W[k, j]); W is [n, n] row-major.
__global__ void cvt_trans_kernel(const float* __restrict__ W, unsigned short* __restrict__ Wt,
                                 int n, int ldout) {
    const int k = blockIdx.x;
    for (int j = threadIdx.x; j < n; j += blockDim.x)
        Wt[(size_t)j * ldout + k] = f2bf(W[(size_t)k * n + j]);
}

// xb[i, 0:254] = bf16(ce[0:254])  (broadcast class-embedding projection)
__global__ void fill_ce_kernel(const float* __restrict__ ce, unsigned short* __restrict__ xb,
                               int ldx, int F) {
    const int i = blockIdx.x;
    for (int j = threadIdx.x; j < F; j += blockDim.x)
        xb[(size_t)i * ldx + j] = f2bf(ce[j]);
}

// ---------------------------------------------------------------------------
// Edge aggregation: agg[dst] += x[src] * norm_src[src]   (bf16 gather, f32 atomics)
// One block per edge, vectorized 2 features / thread.
// ---------------------------------------------------------------------------
__global__ void __launch_bounds__(256)
scatter_feat_kernel(const unsigned short* __restrict__ xb, const int* __restrict__ src,
                    const int* __restrict__ dst, const float* __restrict__ normS,
                    float* __restrict__ agg, int F, int ldx) {
    const int e = blockIdx.x;
    const int s = src[e], d = dst[e];
    const float ns = normS[s];
    const unsigned* xrow = (const unsigned*)(xb + (size_t)s * ldx);
    float* arow = agg + (size_t)d * ldx;
    for (int j2 = threadIdx.x; 2 * j2 < F; j2 += blockDim.x) {
        unsigned p = xrow[j2];
        atomicAdd(arow + 2 * j2,     bf2f((unsigned short)(p & 0xffffu)) * ns);
        atomicAdd(arow + 2 * j2 + 1, bf2f((unsigned short)(p >> 16)) * ns);
    }
}

// Scalar-feature edge aggregation for layer 3.
__global__ void scatter_scalar_kernel(const float* __restrict__ t, const int* __restrict__ src,
                                      const int* __restrict__ dst, const float* __restrict__ normS,
                                      float* __restrict__ oacc, int E) {
    int i = blockIdx.x * blockDim.x + threadIdx.x;
    if (i < E) atomicAdd(&oacc[dst[i]], t[src[i]] * normS[src[i]]);
}

// o[i] = relu(oacc[i] * normD[i] + b3)
__global__ void finish3_kernel(const float* __restrict__ oacc, const float* __restrict__ normD,
                               const float* __restrict__ b3, float* __restrict__ o, int N) {
    int i = blockIdx.x * blockDim.x + threadIdx.x;
    if (i < N) o[i] = fmaxf(oacc[i] * normD[i] + b3[0], 0.0f);
}

// t[i] = dot(xb[i, 0:K], W3[0:K])   (one wave per node, shuffle reduce, wave32)
__global__ void __launch_bounds__(256)
matvec_kernel(const unsigned short* __restrict__ xb, const float* __restrict__ W3,
              float* __restrict__ t, int N, int ldx, int K) {
    const int w = blockIdx.x * (blockDim.x >> 5) + (threadIdx.x >> 5);
    const int lane = threadIdx.x & 31;
    if (w >= N) return;
    const unsigned short* row = xb + (size_t)w * ldx;
    float s = 0.0f;
    for (int k = lane; k < K; k += 32) s += bf2f(row[k]) * W3[k];
    for (int off = 16; off > 0; off >>= 1) s += __shfl_xor(s, off, 32);
    if (lane == 0) t[w] = s;
}

// ---------------------------------------------------------------------------
// WMMA GEMM:  C = act( (A * rowScale) @ B^T + bias )
//   A: [M, lda]  f32 (A_IS_F32) or bf16; M multiple of 16; K multiple of 32.
//   Bt: bf16 [>=Ntiles*16 rows, ldb], Bt[n][k] = B[k][n]  (contiguous K loads)
//   C: bf16 [M, ldc], written at column offset colOff, cols < Nvalid.
// One wave computes one 16x16 tile; tile index is wave-uniform so EXEC stays
// all-ones across every v_wmma (ISA requirement). A re-reads hit 192MB L2.
// ---------------------------------------------------------------------------
template <bool A_IS_F32>
__global__ void __launch_bounds__(256)
gemm_wmma(const void* __restrict__ Aptr, int lda,
          const unsigned short* __restrict__ Bt, int ldb,
          const float* __restrict__ rowScale, const float* __restrict__ bias,
          unsigned short* __restrict__ C, int ldc, int colOff,
          int Mtiles, int Ntiles, int Nvalid, int K, int relu) {
    const int gw = blockIdx.x * (blockDim.x >> 5) + (threadIdx.x >> 5);
    if (gw >= Mtiles * Ntiles) return;                 // wave-uniform exit
    const int tileM = gw / Ntiles;
    const int tileN = gw - tileM * Ntiles;
    const int lane = threadIdx.x & 31;
    const int m  = lane & 15;                          // A row / B col / D col within tile
    const int hi = lane >> 4;                          // half-wave selector

    const int row = tileM * 16 + m;
    float sc = 1.0f;
    if (A_IS_F32 && rowScale) sc = rowScale[row];

    const unsigned short* Brow = Bt + (size_t)(tileN * 16 + m) * ldb;

    v8f acc = {0.f, 0.f, 0.f, 0.f, 0.f, 0.f, 0.f, 0.f};

    for (int k0 = 0; k0 < K; k0 += 32) {
        BfFrag a, b;
        if (A_IS_F32) {
            // A frag: lanes 0-15 hold K = k0+{0..7, 16..23}; lanes 16-31 +8.
            const float* Arow = (const float*)Aptr + (size_t)row * lda + k0 + hi * 8;
#pragma unroll
            for (int g = 0; g < 2; ++g) {
                float4 q0 = ((const float4*)(Arow + g * 16))[0];
                float4 q1 = ((const float4*)(Arow + g * 16))[1];
                a.u[g * 4 + 0] = pk2(q0.x * sc, q0.y * sc);
                a.u[g * 4 + 1] = pk2(q0.z * sc, q0.w * sc);
                a.u[g * 4 + 2] = pk2(q1.x * sc, q1.y * sc);
                a.u[g * 4 + 3] = pk2(q1.z * sc, q1.w * sc);
            }
            if (k0 + 32 < K) __builtin_prefetch(Arow + 32, 0, 1);
        } else {
            const unsigned short* Arow =
                (const unsigned short*)Aptr + (size_t)row * lda + k0 + hi * 8;
            uint4 x0 = ((const uint4*)Arow)[0];
            uint4 x1 = ((const uint4*)(Arow + 16))[0];
            a.u[0] = x0.x; a.u[1] = x0.y; a.u[2] = x0.z; a.u[3] = x0.w;
            a.u[4] = x1.x; a.u[5] = x1.y; a.u[6] = x1.z; a.u[7] = x1.w;
        }
        {
            // B frag: lanes 0-15 hold K = k0+{0..7, 16..23} of column n; lanes 16-31 +8.
            const unsigned short* bp = Brow + k0 + hi * 8;
            uint4 y0 = ((const uint4*)bp)[0];
            uint4 y1 = ((const uint4*)(bp + 16))[0];
            b.u[0] = y0.x; b.u[1] = y0.y; b.u[2] = y0.z; b.u[3] = y0.w;
            b.u[4] = y1.x; b.u[5] = y1.y; b.u[6] = y1.z; b.u[7] = y1.w;
        }
        acc = __builtin_amdgcn_wmma_f32_16x16x32_bf16(
            /*neg_a=*/false, a.v, /*neg_b=*/false, b.v,
            /*c_mod=*/(short)0, acc, /*reuse_a=*/false, /*reuse_b=*/false);
    }

    // Epilogue: D VGPR r holds M = tileM*16 + hi*8 + r, N = tileN*16 + m.
    const int col = tileN * 16 + m;
    if (col < Nvalid) {
        const float bv = bias ? bias[col] : 0.0f;
#pragma unroll
        for (int r = 0; r < 8; ++r) {
            float v = acc[r] + bv;
            if (relu) v = fmaxf(v, 0.0f);
            const int rr = tileM * 16 + hi * 8 + r;
            C[(size_t)rr * ldc + colOff + col] = f2bf(v);
        }
    }
}

// ---------------------------------------------------------------------------
// Launcher
// ---------------------------------------------------------------------------
extern "C" void kernel_launch(void* const* d_in, const int* in_sizes, int n_in,
                              void* d_out, int out_size, void* d_ws, size_t ws_size,
                              hipStream_t stream) {
    const float* class_embed = (const float*)d_in[0];
    const float* all_glove   = (const float*)d_in[1];
    const float* W_word = (const float*)d_in[2];
    const float* b_word = (const float*)d_in[3];
    const float* W_img  = (const float*)d_in[4];
    const float* b_img  = (const float*)d_in[5];
    const float* W1 = (const float*)d_in[6];
    const float* b1 = (const float*)d_in[7];
    const float* W2 = (const float*)d_in[8];
    const float* b2 = (const float*)d_in[9];
    const float* W3 = (const float*)d_in[10];
    const float* b3 = (const float*)d_in[11];
    const float* W_fin = (const float*)d_in[12];
    const float* b_fin = (const float*)d_in[13];
    const int* src = (const int*)d_in[14];
    const int* dst = (const int*)d_in[15];

    const int N = in_sizes[0];          // 20000 (multiple of 16)
    const int E = in_sizes[14];         // 660000
    const int GLOVE = 300, GLOVE_P = 320;
    const int F = 254;                  // feature half-dim / output dim
    const int IN = 508, IN_P = 512;

    // Workspace carve-up (256B aligned), ~76 MB total.
    char* base = (char*)d_ws;
    size_t off = 0;
    auto carve = [&](size_t bytes) -> char* {
        char* p = base + off;
        off += (bytes + 255) & ~(size_t)255;
        return p;
    };
    float* normS = (float*)carve((size_t)N * 4);
    float* normD = (float*)carve((size_t)N * 4);
    float* ce    = (float*)carve(256 * 4);
    float* tvec  = (float*)carve((size_t)N * 4);
    float* oacc  = (float*)carve((size_t)N * 4);
    float* ovec  = (float*)carve((size_t)N * 4);
    unsigned short* gloveb = (unsigned short*)carve((size_t)N * GLOVE_P * 2);
    unsigned short* wordb  = (unsigned short*)carve((size_t)256 * GLOVE_P * 2);
    unsigned short* w1t    = (unsigned short*)carve((size_t)IN_P * IN_P * 2);
    unsigned short* w2t    = (unsigned short*)carve((size_t)IN_P * IN_P * 2);
    unsigned short* xb     = (unsigned short*)carve((size_t)N * IN_P * 2);
    float* agg             = (float*)carve((size_t)N * IN_P * 4);

    // Zero everything that is read-with-padding or atomically accumulated.
    hipMemsetAsync(normS, 0, (size_t)N * 4, stream);
    hipMemsetAsync(normD, 0, (size_t)N * 4, stream);
    hipMemsetAsync(gloveb, 0, (size_t)N * GLOVE_P * 2, stream);
    hipMemsetAsync(wordb, 0, (size_t)256 * GLOVE_P * 2, stream);
    hipMemsetAsync(w1t, 0, (size_t)IN_P * IN_P * 2, stream);
    hipMemsetAsync(w2t, 0, (size_t)IN_P * IN_P * 2, stream);
    hipMemsetAsync(xb, 0, (size_t)N * IN_P * 2, stream);
    hipMemsetAsync(agg, 0, (size_t)N * IN_P * 4, stream);
    hipMemsetAsync(oacc, 0, (size_t)N * 4, stream);

    // Degrees -> norms.
    deg_kernel<<<(E + 255) / 256, 256, 0, stream>>>(src, dst, normS, normD, E);
    norm_kernel<<<(N + 255) / 256, 256, 0, stream>>>(normS, normD, N);

    // ce = class_embed @ W_img.T + b_img   [254]
    vecmat_kernel<<<F, 256, 0, stream>>>(class_embed, W_img, b_img, ce, N);

    // bf16 conversions (padded K): glove, W_word (already [out,in]), W1/W2 transposed.
    cvt_pad_kernel<<<N, 256, 0, stream>>>(all_glove, gloveb, GLOVE, GLOVE_P);
    cvt_pad_kernel<<<F, 256, 0, stream>>>(W_word, wordb, GLOVE, GLOVE_P);
    cvt_trans_kernel<<<IN, 256, 0, stream>>>(W1, w1t, IN, IN_P);
    cvt_trans_kernel<<<IN, 256, 0, stream>>>(W2, w2t, IN, IN_P);

    // we = glove @ W_word.T + b_word -> xb[:, 254:508] (bf16)
    {
        const int Mtiles = N / 16, Ntiles = 256 / 16;
        const int waves = Mtiles * Ntiles;
        gemm_wmma<false><<<(waves + 7) / 8, 256, 0, stream>>>(
            gloveb, GLOVE_P, wordb, GLOVE_P, nullptr, b_word,
            xb, IN_P, /*colOff=*/F, Mtiles, Ntiles, /*Nvalid=*/F, GLOVE_P, /*relu=*/0);
    }
    // xb[:, 0:254] = broadcast(ce)
    fill_ce_kernel<<<N, 256, 0, stream>>>(ce, xb, IN_P, F);

    // ---- Layer 1: agg = scatter(x * nS); h1 = relu((agg * nD) @ W1 + b1) ----
    scatter_feat_kernel<<<E, 256, 0, stream>>>(xb, src, dst, normS, agg, IN, IN_P);
    {
        const int Mtiles = N / 16, Ntiles = IN_P / 16;
        const int waves = Mtiles * Ntiles;
        gemm_wmma<true><<<(waves + 7) / 8, 256, 0, stream>>>(
            agg, IN_P, w1t, IN_P, normD, b1,
            xb, IN_P, 0, Mtiles, Ntiles, IN, IN_P, /*relu=*/1);
    }

    // ---- Layer 2 ----
    hipMemsetAsync(agg, 0, (size_t)N * IN_P * 4, stream);
    scatter_feat_kernel<<<E, 256, 0, stream>>>(xb, src, dst, normS, agg, IN, IN_P);
    {
        const int Mtiles = N / 16, Ntiles = IN_P / 16;
        const int waves = Mtiles * Ntiles;
        gemm_wmma<true><<<(waves + 7) / 8, 256, 0, stream>>>(
            agg, IN_P, w2t, IN_P, normD, b2,
            xb, IN_P, 0, Mtiles, Ntiles, IN, IN_P, /*relu=*/1);
    }

    // ---- Layer 3 (pre-mult, F=1): t = h2 @ W3; o = relu(scatter(t*nS)*nD + b3) ----
    matvec_kernel<<<(N + 7) / 8, 256, 0, stream>>>(xb, W3, tvec, N, IN_P, IN);
    scatter_scalar_kernel<<<(E + 255) / 256, 256, 0, stream>>>(tvec, src, dst, normS, oacc, E);
    finish3_kernel<<<(N + 255) / 256, 256, 0, stream>>>(oacc, normD, b3, ovec, N);

    // out = o @ W_fin.T + b_fin   [254] f32
    vecmat_kernel<<<F, 256, 0, stream>>>(ovec, W_fin, b_fin, (float*)d_out, N);
}
